// AutoregressiveModel_71571335020655
// MI455X (gfx1250) — matrix operand order repaired and checked by reading, not compiled, not run on hardware
//
#include <hip/hip_runtime.h>
#include <hip/hip_bf16.h>

#define UNITS 4096
#define BATCH 256
#define DEPTH 13

typedef __attribute__((ext_vector_type(16))) _Float16 v16h;
typedef __attribute__((ext_vector_type(8)))  _Float16 v8h;
typedef __attribute__((ext_vector_type(8)))  float    v8f;

// ---------------------------------------------------------------------------
// Prep: convert W2 (13 x 64 x 64 f32, row-major [k][o][i]) to f16 in workspace
// ---------------------------------------------------------------------------
__global__ void hl_convert_w2(const float* __restrict__ w, _Float16* __restrict__ o, int n) {
    int i = blockIdx.x * 256 + threadIdx.x;
    if (i < n) o[i] = (_Float16)w[i];
}

// ---------------------------------------------------------------------------
// Layer 1: h1[b,j] = relu( sum_{k=1..L(j)} W1[k-1] @ x[b, j>>k] + b1[k-1] )
// in_f = 2 -> plain VALU f32; output a1 stored f16 as [node][batch][64]
// thread = (oq:2 | b:8 | j:12), oq selects 16 of the 64 outputs
// ---------------------------------------------------------------------------
__global__ void hl_layer1(const float* __restrict__ x, const float* __restrict__ W1,
                          const float* __restrict__ b1, _Float16* __restrict__ a1) {
    unsigned tid = blockIdx.x * 256u + threadIdx.x;
    int oq = tid & 3;
    int b  = (tid >> 2) & 255;
    int j  = tid >> 10;
    float acc[16];
#pragma unroll
    for (int o = 0; o < 16; ++o) acc[o] = 0.f;
    for (int k = 1; k < DEPTH; ++k) {
        if ((j >> (k - 1)) == 0) break;      // validity is monotone in k
        int n = j >> k;
        float x0 = x[((size_t)b * UNITS + n) * 2 + 0];
        float x1 = x[((size_t)b * UNITS + n) * 2 + 1];
        const float* w  = W1 + ((size_t)(k - 1) * 64 + oq * 16) * 2;
        const float* bb = b1 + (size_t)(k - 1) * 64 + oq * 16;
#pragma unroll
        for (int o = 0; o < 16; ++o)
            acc[o] += w[o * 2] * x0 + w[o * 2 + 1] * x1 + bb[o];
    }
    v8h* dst = (v8h*)(a1 + ((size_t)j * BATCH + b) * 64 + oq * 16);
    v8h r0, r1;
#pragma unroll
    for (int o = 0; o < 8; ++o) {
        r0[o] = (_Float16)fmaxf(acc[o],     0.f);
        r1[o] = (_Float16)fmaxf(acc[o + 8], 0.f);
    }
    dst[0] = r0;
    dst[1] = r1;
}

// ---------------------------------------------------------------------------
// Layer 2: WMMA. act layout [node][batch][64] f16.
// A fragment (16x32 f16): lane l = row (l&15); elems 0..7 = K base+(l>>4)*8,
// elems 8..15 = K base+16+(l>>4)*8   (per CDNA5 16-bit A-matrix layout)
// ---------------------------------------------------------------------------
__device__ __forceinline__ v16h hl_load_a(const _Float16* __restrict__ act,
                                          int node, int b0, int lane, int kfrag) {
    const _Float16* row = act + ((size_t)node * BATCH + b0 + (lane & 15)) * 64;
    const _Float16* p   = row + kfrag * 32 + ((lane >> 4) & 1) * 8;
    v8h lo = *(const v8h*)p;
    v8h hi = *(const v8h*)(p + 16);
    v16h r;
#pragma unroll
    for (int i = 0; i < 8; ++i) { r[i] = lo[i]; r[i + 8] = hi[i]; }
    return r;
}

// B fragment (32x16 f16): B = W^T, so lane l (col o = ot*16 + (l&15)) reads W row o:
// elems 0..7 = K kfrag*32+(l>>4)*8, elems 8..15 = K kfrag*32+16+(l>>4)*8.
__device__ __forceinline__ v16h hl_load_b(const _Float16* __restrict__ Wk,
                                          int ot, int kfrag, int lane) {
    const _Float16* p = Wk + (size_t)(ot * 16 + (lane & 15)) * 64
                           + kfrag * 32 + ((lane >> 4) & 1) * 8;
    v8h lo = *(const v8h*)p;
    v8h hi = *(const v8h*)(p + 16);
    v16h r;
#pragma unroll
    for (int i = 0; i < 8; ++i) { r[i] = lo[i]; r[i + 8] = hi[i]; }
    return r;
}

// One GEMM unit (NO bias): acc(16x64,f32) += act[16 batches, node] @ W[k]^T
// = 2 K-frags x 4 N-tiles = 8 x v_wmma_f32_16x16x32_f16, pure WMMA chain
__device__ __forceinline__ void hl_gemm_unit(v8f acc[4], const _Float16* __restrict__ act,
                                             int node, int b0,
                                             const _Float16* __restrict__ Wk, int lane) {
    v16h af0 = hl_load_a(act, node, b0, lane, 0);
    v16h af1 = hl_load_a(act, node, b0, lane, 1);
#pragma unroll
    for (int ot = 0; ot < 4; ++ot) {
        v16h bf0 = hl_load_b(Wk, ot, 0, lane);
        v16h bf1 = hl_load_b(Wk, ot, 1, lane);
        acc[ot] = __builtin_amdgcn_wmma_f32_16x16x32_f16(false, af0, false, bf0,
                                                         (short)0, acc[ot], false, false);
        acc[ot] = __builtin_amdgcn_wmma_f32_16x16x32_f16(false, af1, false, bf1,
                                                         (short)0, acc[ot], false, false);
    }
}

// store relu(acc) as f16 rows of a2 (C/D layout: M = vgpr + (lane>>4)*8, N = lane&15)
__device__ __forceinline__ void hl_store_tile(v8f acc[4], _Float16* __restrict__ a2,
                                              int j, int b0, int lane) {
    int nl = lane & 15;
    int hb = ((lane >> 4) & 1) * 8;
#pragma unroll
    for (int ot = 0; ot < 4; ++ot)
#pragma unroll
        for (int r = 0; r < 8; ++r) {
            float v = fmaxf(acc[ot][r], 0.f);
            a2[((size_t)j * BATCH + b0 + hb + r) * 64 + ot * 16 + nl] = (_Float16)v;
        }
}

// ONE WAVE PER WORKGROUP: task = blockIdx.x is an SGPR, so tile id t, level L
// and every loop bound below are provably scalar -> pure s_cbranch control
// flow, EXEC stays all-ones around every WMMA (ISA requirement), and no
// per-lane mask bookkeeping is emitted.
//
// wave-task = (batch-group bg of 16, node tile t of 16 aligned nodes)
// t>=1: all 16 nodes share level L = msb(16t)+1; k>=4 terms are tile-constant
// -> C = sum_{k=4..L} y_k(t >> (k-4)) + SUM of ALL biases (k=0..L, same valid
//    set for every child); then depth-first descent P_d = P_{d+1} + y_d(j>>d)
//    with bias-free units so WMMAs chain through the accumulator.
__global__ void __launch_bounds__(32) hl_layer2(const _Float16* __restrict__ a1,
                                                const _Float16* __restrict__ W2h,
                                                const float* __restrict__ b2,
                                                _Float16* __restrict__ a2) {
    int lane = threadIdx.x;                    // 0..31
    int task = blockIdx.x;                     // scalar: 4096 tasks
    int bg   = task >> 8;                      // 0..15
    int t    = task & 255;                     // node tile (16 nodes)
    int b0   = bg * 16;
    int nl   = lane & 15;

    if (t == 0) {                              // tile crosses heap levels: generic path
        for (int j = 0; j < 16; ++j) {
            v8f acc[4] = {};
            float bsum[4] = {0.f, 0.f, 0.f, 0.f};
            for (int k = 0; k < DEPTH; ++k) {
                if (k > 0 && (j >> (k - 1)) == 0) break;
                hl_gemm_unit(acc, a1, j >> k, b0, W2h + (size_t)k * 4096, lane);
#pragma unroll
                for (int ot = 0; ot < 4; ++ot) bsum[ot] += b2[k * 64 + ot * 16 + nl];
            }
#pragma unroll
            for (int ot = 0; ot < 4; ++ot)
#pragma unroll
                for (int r = 0; r < 8; ++r) acc[ot][r] += bsum[ot];
            hl_store_tile(acc, a2, j, b0, lane);
        }
        return;
    }

    int j0 = t * 16;
    int L  = 32 - __builtin_clz((unsigned)j0);     // msb(j0)+1, in [5,12]

    v8f C[4] = {};
    float cb[4];
#pragma unroll
    for (int ot = 0; ot < 4; ++ot)                 // biases of always-valid k=0..3
        cb[ot] = b2[0 * 64 + ot * 16 + nl] + b2[1 * 64 + ot * 16 + nl]
               + b2[2 * 64 + ot * 16 + nl] + b2[3 * 64 + ot * 16 + nl];
    for (int k = 4; k <= L; ++k) {                 // tile-constant ancestor terms
        hl_gemm_unit(C, a1, t >> (k - 4), b0, W2h + (size_t)k * 4096, lane);
#pragma unroll
        for (int ot = 0; ot < 4; ++ot) cb[ot] += b2[k * 64 + ot * 16 + nl];
    }
#pragma unroll
    for (int ot = 0; ot < 4; ++ot)                 // all biases folded in ONCE
#pragma unroll
        for (int r = 0; r < 8; ++r) C[ot][r] += cb[ot];

    for (int m3 = 0; m3 < 2; ++m3) {
        v8f P3[4] = {C[0], C[1], C[2], C[3]};
        int n3 = 2 * t + m3;
        hl_gemm_unit(P3, a1, n3, b0, W2h + (size_t)3 * 4096, lane);
        for (int m2 = 0; m2 < 2; ++m2) {
            v8f P2[4] = {P3[0], P3[1], P3[2], P3[3]};
            int n2 = 2 * n3 + m2;
            hl_gemm_unit(P2, a1, n2, b0, W2h + (size_t)2 * 4096, lane);
            for (int m1 = 0; m1 < 2; ++m1) {
                v8f P1[4] = {P2[0], P2[1], P2[2], P2[3]};
                int n1 = 2 * n2 + m1;
                hl_gemm_unit(P1, a1, n1, b0, W2h + (size_t)1 * 4096, lane);
                for (int m0 = 0; m0 < 2; ++m0) {
                    v8f P0[4] = {P1[0], P1[1], P1[2], P1[3]};
                    int n0 = 2 * n1 + m0;
                    hl_gemm_unit(P0, a1, n0, b0, W2h, lane);
                    hl_store_tile(P0, a2, n0, b0, lane);
                }
            }
        }
    }
}

// ---------------------------------------------------------------------------
// Layer 3: out[b,j,0:2] = sum_k W3[k](2x64) @ a2[b, j>>k] + b3[k]
// thread = (b:8 | j:12); writes are 8B contiguous across consecutive j.
// Activation rows read as 16B v8h vectors, weights as float4.
// ---------------------------------------------------------------------------
__global__ void hl_layer3(const _Float16* __restrict__ a2, const float* __restrict__ W3,
                          const float* __restrict__ b3, float* __restrict__ out) {
    unsigned tid = blockIdx.x * 256u + threadIdx.x;
    int j = tid & (UNITS - 1);
    int b = tid >> 12;
    float acc0 = 0.f, acc1 = 0.f;
    for (int k = 0; k < DEPTH; ++k) {
        if (k > 0 && (j >> (k - 1)) == 0) break;
        const v8h*    h  = (const v8h*)(a2 + ((size_t)(j >> k) * BATCH + b) * 64);
        const float4* w0 = (const float4*)(W3 + (size_t)k * 128);
        const float4* w1 = (const float4*)(W3 + (size_t)k * 128 + 64);
        float s0 = 0.f, s1 = 0.f;
#pragma unroll
        for (int c = 0; c < 8; ++c) {
            v8h hh = h[c];
            float4 a = w0[c * 2 + 0], bq = w0[c * 2 + 1];
            float4 cc = w1[c * 2 + 0], dq = w1[c * 2 + 1];
            s0 += a.x  * (float)hh[0] + a.y  * (float)hh[1] + a.z  * (float)hh[2] + a.w  * (float)hh[3];
            s0 += bq.x * (float)hh[4] + bq.y * (float)hh[5] + bq.z * (float)hh[6] + bq.w * (float)hh[7];
            s1 += cc.x * (float)hh[0] + cc.y * (float)hh[1] + cc.z * (float)hh[2] + cc.w * (float)hh[3];
            s1 += dq.x * (float)hh[4] + dq.y * (float)hh[5] + dq.z * (float)hh[6] + dq.w * (float)hh[7];
        }
        acc0 += s0 + b3[k * 2 + 0];
        acc1 += s1 + b3[k * 2 + 1];
    }
    out[((size_t)b * UNITS + j) * 2 + 0] = acc0;
    out[((size_t)b * UNITS + j) * 2 + 1] = acc1;
}

// ---------------------------------------------------------------------------
extern "C" void kernel_launch(void* const* d_in, const int* in_sizes, int n_in,
                              void* d_out, int out_size, void* d_ws, size_t ws_size,
                              hipStream_t stream) {
    const float* x  = (const float*)d_in[0];
    const float* W1 = (const float*)d_in[1];
    const float* b1 = (const float*)d_in[2];
    const float* W2 = (const float*)d_in[3];
    const float* b2 = (const float*)d_in[4];
    const float* W3 = (const float*)d_in[5];
    const float* b3 = (const float*)d_in[6];
    float* out = (float*)d_out;

    const size_t act_elems = (size_t)UNITS * BATCH * 64;     // 67,108,864 f16
    _Float16* a1  = (_Float16*)d_ws;
    _Float16* a2  = a1 + act_elems;
    _Float16* W2h = a2 + act_elems;
    const int w2n = DEPTH * 64 * 64;                         // 53,248

    hl_convert_w2<<<(w2n + 255) / 256, 256, 0, stream>>>(W2, W2h, w2n);
    hl_layer1<<<(UNITS * BATCH * 4) / 256, 256, 0, stream>>>(x, W1, b1, a1);
    hl_layer2<<<16 * 256, 32, 0, stream>>>(a1, W2h, b2, a2);
    hl_layer3<<<(UNITS * BATCH) / 256, 256, 0, stream>>>(a2, W3, b3, out);
}
